// AsteroidSurrogate_75496935129488
// MI455X (gfx1250) — matrix operand order repaired and checked by reading, compiled, not run on hardware
//
#include <hip/hip_runtime.h>
#include <math.h>

#define B_    4096
#define T_    128
#define P_    4
#define F_    30
#define GH_   64
#define IH_   128
#define LH_   128
#define MT_   16      // batch rows per workgroup
#define NTH_  256     // 8 wave32

typedef __attribute__((ext_vector_type(16))) __bf16 v16bf;
typedef __attribute__((ext_vector_type(8)))  float  v8f;
typedef __attribute__((ext_vector_type(4)))  float  vf4;
typedef __attribute__((ext_vector_type(2)))  float  v2f;

union FragBF { v16bf v; vf4 f4[2]; unsigned short u[16]; };

// ---------------- LDS layout (bytes), total 326144 <= 327680 (320KB/WGP) ---
#define OFF_WIH   0u        // [512][128] bf16  (lstm_Wih, already [N][K])
#define OFF_WHH   131072u   // [512][128] bf16
#define OFF_W1A   262144u   // [128][64]  bf16  (inp_W1 agg part, transposed)
#define OFF_W1S   278528u   // [4][128]   f32   (inp_W1 state part, [K][N])
#define OFF_GW1   280576u   // [4][64]    f32   (gnn_W1 [K][N])
#define OFF_GW2   281600u   // [64][64]   bf16  (gnn_W2 transposed)
#define OFF_HW2   289792u   // [128][4]   f32   (head_W2 [K][N])
#define OFF_B     291840u   // 1600 f32 biases
#define OFF_FEAT  298240u   // [64][4]    f32   GNN features (planet-major rows)
#define OFF_AGG   299264u   // [16][64]   f32   planet-summed GNN output
#define OFF_AGGB  303360u   // [16][64]   bf16
#define OFF_G1    305408u   // [64][64]   bf16  GNN hidden
#define OFF_T0    313600u   // [16][128]  bf16  MLP hidden staging
#define OFF_XT    317696u   // [16][128]  bf16  LSTM input x_t
#define OFF_H     321792u   // [16][128]  bf16  LSTM hidden h
#define OFF_STATE 325888u   // [16][4]    f32   asteroid state
#define SMEM_TOTAL 326144u
static_assert(SMEM_TOTAL <= 320u * 1024u, "LDS budget exceeded");

// bias sub-offsets (floats) inside OFF_B
#define BO_GB1  0
#define BO_GB2  64
#define BO_IB1  128
#define BO_IB2  256
#define BO_BIH  384
#define BO_BHH  896
#define BO_HB1  1408
#define BO_HB2  1536

__device__ __forceinline__ unsigned short f2bf(float x) {
  unsigned int u = __float_as_uint(x);
  u += 0x7fffu + ((u >> 16) & 1u);          // round-to-nearest-even
  return (unsigned short)(u >> 16);
}
__device__ __forceinline__ float bf2f(unsigned short h) {
  return __uint_as_float(((unsigned int)h) << 16);
}
__device__ __forceinline__ float sigm(float x) { return 1.0f / (1.0f + __expf(-x)); }
// gfx1250 has a hardware v_tanh_f32 (TRANS op) — use it if the builtin exists.
__device__ __forceinline__ float tanh_fast(float x) {
#if __has_builtin(__builtin_amdgcn_tanhf)
  return __builtin_amdgcn_tanhf(x);
#elif __has_builtin(__builtin_amdgcn_tanh_f32)
  return __builtin_amdgcn_tanh_f32(x);
#else
  float e = __expf(-2.0f * x);              // 4 VALU instead of OCML tanh
  return (1.0f - e) * __frcp_rn(1.0f + e);
#endif
}
__device__ __forceinline__ float relu(float x) { return fmaxf(x, 0.0f); }

// ---- WMMA wrappers (CDNA5 gfx1250) --------------------------------------
__device__ __forceinline__ v8f wmma_bf16(v16bf a, v16bf b, v8f c) {
  return __builtin_amdgcn_wmma_f32_16x16x32_bf16(false, a, false, b, (short)0, c,
                                                 false, false);
}
__device__ __forceinline__ v8f wmma_f32k4(v2f a, v2f b, v8f c) {
  return __builtin_amdgcn_wmma_f32_16x16x4_f32(false, a, false, b, (short)0, c,
                                               false, false);
}

// A fragment: 16xK slab, row-major bf16, K contiguous (contiguous layout:
// lanes<16 hold K {k0..k0+15}; lanes>=16 hold K {k0+16..k0+31})
__device__ __forceinline__ v16bf ldA(const unsigned short* base, int ldk, int k0,
                                     int lane) {
  int m = lane & 15;
  int kb = k0 + ((lane & 16) ? 16 : 0);
  FragBF f;
  const unsigned short* p = base + m * ldk + kb;
  f.f4[0] = *(const vf4*)(p);
  f.f4[1] = *(const vf4*)(p + 8);
  return f.v;
}
// B fragment: weights stored [N][K]; each lane reads 16 contiguous K for one n
__device__ __forceinline__ v16bf ldB(const unsigned short* base, int ldk, int n0,
                                     int k0, int lane) {
  int n = n0 + (lane & 15);
  int kb = k0 + ((lane & 16) ? 16 : 0);
  FragBF f;
  const unsigned short* p = base + n * ldk + kb;
  f.f4[0] = *(const vf4*)(p);
  f.f4[1] = *(const vf4*)(p + 8);
  return f.v;
}

struct Ctx {
  const unsigned short* sWih; const unsigned short* sWhh;
  const unsigned short* sW1A; const float* sW1S;
  const float* sGW1;          const unsigned short* sGW2;
  const float* sHW2;          float* sB;
  float* sFeat; float* sAgg;  unsigned short* sAggB;
  unsigned short* sG1; unsigned short* sT0; unsigned short* sXt;
  unsigned short* sH;  float* sState;
  const unsigned short* wsInpT; const unsigned short* wsHeadT;
  const float* masses;
  int tid, wave, lane;
};

// GNN (4 planets -> 64) + input MLP (68 -> 128 -> 128), writes x_t to sXt
__device__ __forceinline__ void gnn_inp(const Ctx& c, const float* pb, int pstride) {
  const v8f Z8 = {0.f,0.f,0.f,0.f,0.f,0.f,0.f,0.f};
  // features: row r = p*16 + m : [dx, dy, 1/|d|, mass_p]
  if (c.tid < 64) {
    int p = c.tid >> 4, m = c.tid & 15;
    float ax = c.sState[m * 4 + 0], ay = c.sState[m * 4 + 1];
    const float* pp = pb + (long)m * pstride + p * 2;
    __builtin_prefetch(pp + P_ * 2, 0, 0);   // next timestep's planet row
    float dx = pp[0] - ax, dy = pp[1] - ay;
    float inv = rsqrtf(dx * dx + dy * dy + 1e-6f);
    int r = p * 16 + m;
    c.sFeat[r*4+0] = dx; c.sFeat[r*4+1] = dy;
    c.sFeat[r*4+2] = inv; c.sFeat[r*4+3] = c.masses[p];
  }
  for (int i = c.tid; i < MT_ * GH_; i += NTH_) c.sAgg[i] = 0.0f;
  __syncthreads();
  // GNN layer1: 64x64 = 4x4 tiles, 2 per wave, K=4 in fp32 WMMA
#pragma unroll
  for (int tt = 0; tt < 2; ++tt) {
    int ti = c.wave * 2 + tt, rt = ti >> 2, ct = ti & 3;
    int kb = (c.lane & 16) ? 2 : 0;
    int m  = rt * 16 + (c.lane & 15);
    int n  = ct * 16 + (c.lane & 15);
    v2f a, b;
    a.x = c.sFeat[m * 4 + kb];        a.y = c.sFeat[m * 4 + kb + 1];
    b.x = c.sGW1[kb * GH_ + n];       b.y = c.sGW1[(kb + 1) * GH_ + n];
    v8f acc = wmma_f32k4(a, b, Z8);
    float bb = c.sB[BO_GB1 + n];
    int r0 = ((c.lane & 16) ? 8 : 0) + rt * 16;
#pragma unroll
    for (int v = 0; v < 8; ++v)
      c.sG1[(r0 + v) * GH_ + n] = f2bf(relu(acc[v] + bb));
  }
  __syncthreads();
  // GNN layer2 (64x64, K=64) + sum over planets via ds_add_f32
  {
    int ct = c.wave & 3, rt0 = c.wave >> 2;
    v8f accs[2];
#pragma unroll
    for (int rr = 0; rr < 2; ++rr) {
      int rt = rt0 + rr * 2;
      v8f acc = Z8;
#pragma unroll
      for (int k0 = 0; k0 < GH_; k0 += 32)
        acc = wmma_bf16(ldA(c.sG1 + rt * 16 * GH_, GH_, k0, c.lane),
                        ldB(c.sGW2, GH_, ct * 16, k0, c.lane), acc);
      accs[rr] = acc;
    }
    int n = ct * 16 + (c.lane & 15);
    float bb = c.sB[BO_GB2 + n];
    int r0 = (c.lane & 16) ? 8 : 0;
#pragma unroll
    for (int v = 0; v < 8; ++v) {
      float s = relu(accs[0][v] + bb) + relu(accs[1][v] + bb);
      atomicAdd(&c.sAgg[(r0 + v) * GH_ + n], s);         // planet reduce
    }
  }
  __syncthreads();
  for (int i = c.tid; i < MT_ * GH_; i += NTH_) c.sAggB[i] = f2bf(c.sAgg[i]);
  __syncthreads();
  // input MLP layer1: K = 4 (state, fp32 WMMA) + 64 (agg, bf16 WMMA)
  {
    int n0 = c.wave * 16;
    int kb = (c.lane & 16) ? 2 : 0;
    int m  = c.lane & 15, n = n0 + (c.lane & 15);
    v2f a, b;
    a.x = c.sState[m * 4 + kb];       a.y = c.sState[m * 4 + kb + 1];
    b.x = c.sW1S[kb * IH_ + n];       b.y = c.sW1S[(kb + 1) * IH_ + n];
    v8f acc = wmma_f32k4(a, b, Z8);
#pragma unroll
    for (int k0 = 0; k0 < GH_; k0 += 32)
      acc = wmma_bf16(ldA(c.sAggB, GH_, k0, c.lane),
                      ldB(c.sW1A, GH_, n0, k0, c.lane), acc);
    float bb = c.sB[BO_IB1 + n];
    int r0 = (c.lane & 16) ? 8 : 0;
#pragma unroll
    for (int v = 0; v < 8; ++v)
      c.sT0[(r0 + v) * IH_ + n] = f2bf(relu(acc[v] + bb));
  }
  __syncthreads();
  // input MLP layer2: K=128, B streamed from L2-resident transposed copy
  {
    int n0 = c.wave * 16;
    v8f acc = Z8;
#pragma unroll
    for (int k0 = 0; k0 < IH_; k0 += 32)
      acc = wmma_bf16(ldA(c.sT0, IH_, k0, c.lane),
                      ldB(c.wsInpT, IH_, n0, k0, c.lane), acc);
    int n = n0 + (c.lane & 15);
    float bb = c.sB[BO_IB2 + n];
    int r0 = (c.lane & 16) ? 8 : 0;
#pragma unroll
    for (int v = 0; v < 8; ++v)
      c.sXt[(r0 + v) * IH_ + n] = f2bf(relu(acc[v] + bb));
  }
  __syncthreads();
}

// LSTM cell: wave w owns columns w*16..w*16+15 of all 4 gates; c stays in regs.
// Half of the Whh B-fragments (K-blocks 0,1) are register-resident (64 VGPRs);
// the rest stream from LDS — keeps the wave under 256 VGPRs so 2 waves/SIMD fit.
__device__ __forceinline__ void lstm_step(const Ctx& c, v8f& creg,
                                          const v16bf* whhF) {
  const v8f Z8 = {0.f,0.f,0.f,0.f,0.f,0.f,0.f,0.f};
  v8f acc[4] = {Z8, Z8, Z8, Z8};
#pragma unroll
  for (int kk = 0; kk < 4; ++kk) {
    int k0 = kk * 32;
    v16bf ax = ldA(c.sXt, IH_, k0, c.lane);
#pragma unroll
    for (int g = 0; g < 4; ++g)
      acc[g] = wmma_bf16(ax, ldB(c.sWih, IH_, g * LH_ + c.wave * 16, k0, c.lane),
                         acc[g]);
    v16bf ah = ldA(c.sH, LH_, k0, c.lane);
#pragma unroll
    for (int g = 0; g < 4; ++g) {
      v16bf bw = (kk < 2) ? whhF[kk * 4 + g]
                          : ldB(c.sWhh, LH_, g * LH_ + c.wave * 16, k0, c.lane);
      acc[g] = wmma_bf16(ah, bw, acc[g]);
    }
  }
  __syncthreads();                       // all reads of old sH done
  int col = c.wave * 16 + (c.lane & 15);
  float bi = c.sB[BO_BIH + 0*LH_ + col] + c.sB[BO_BHH + 0*LH_ + col];
  float bf = c.sB[BO_BIH + 1*LH_ + col] + c.sB[BO_BHH + 1*LH_ + col];
  float bg = c.sB[BO_BIH + 2*LH_ + col] + c.sB[BO_BHH + 2*LH_ + col];
  float bo = c.sB[BO_BIH + 3*LH_ + col] + c.sB[BO_BHH + 3*LH_ + col];
  int r0 = (c.lane & 16) ? 8 : 0;
#pragma unroll
  for (int v = 0; v < 8; ++v) {
    float cc = sigm(acc[1][v] + bf) * creg[v] +
               sigm(acc[0][v] + bi) * tanh_fast(acc[2][v] + bg);
    creg[v] = cc;
    float hh = sigm(acc[3][v] + bo) * tanh_fast(cc);
    c.sH[(r0 + v) * LH_ + col] = f2bf(hh);
  }
  __syncthreads();
}

// pre-transpose inp_W2 and head_W1 to bf16 [N][K] in d_ws (L2-resident)
__global__ void asteroid_pack_weights(const float* __restrict__ iW2,
                                      const float* __restrict__ hW1,
                                      unsigned short* __restrict__ wsInp,
                                      unsigned short* __restrict__ wsHead) {
  int i = blockIdx.x * blockDim.x + threadIdx.x;
  if (i < IH_ * IH_) {
    int n = i >> 7, k = i & 127;
    wsInp[n * IH_ + k]  = f2bf(iW2[k * IH_ + n]);
    wsHead[n * IH_ + k] = f2bf(hW1[k * IH_ + n]);
  }
}

__global__ __launch_bounds__(NTH_)
__attribute__((amdgpu_waves_per_eu(2)))   // guarantee 2 waves/SIMD => 8-wave WG fits
void asteroid_main(
    const float* __restrict__ pastP, const float* __restrict__ pastS,
    const float* __restrict__ futP,  const float* __restrict__ masses,
    const float* __restrict__ gW1, const float* __restrict__ gb1,
    const float* __restrict__ gW2, const float* __restrict__ gb2,
    const float* __restrict__ iW1, const float* __restrict__ ib1,
    const float* __restrict__ ib2,
    const float* __restrict__ Wih, const float* __restrict__ Whh,
    const float* __restrict__ bih, const float* __restrict__ bhh,
    const float* __restrict__ hb1, const float* __restrict__ hW2,
    const float* __restrict__ hb2,
    const unsigned short* __restrict__ wsInpT,
    const unsigned short* __restrict__ wsHeadT,
    float* __restrict__ out) {
  extern __shared__ char smem[];
  const int tid = threadIdx.x, wave = tid >> 5, lane = tid & 31;
  const int b0 = blockIdx.x * MT_;
  const v8f Z8 = {0.f,0.f,0.f,0.f,0.f,0.f,0.f,0.f};

  Ctx c;
  unsigned short* sWih = (unsigned short*)(smem + OFF_WIH);
  unsigned short* sWhh = (unsigned short*)(smem + OFF_WHH);
  unsigned short* sW1A = (unsigned short*)(smem + OFF_W1A);
  float*          sW1S = (float*)(smem + OFF_W1S);
  float*          sGW1 = (float*)(smem + OFF_GW1);
  unsigned short* sGW2 = (unsigned short*)(smem + OFF_GW2);
  float*          sHW2 = (float*)(smem + OFF_HW2);
  float*          sB   = (float*)(smem + OFF_B);
  c.sWih = sWih; c.sWhh = sWhh; c.sW1A = sW1A; c.sW1S = sW1S;
  c.sGW1 = sGW1; c.sGW2 = sGW2; c.sHW2 = sHW2; c.sB = sB;
  c.sFeat = (float*)(smem + OFF_FEAT);
  c.sAgg  = (float*)(smem + OFF_AGG);
  c.sAggB = (unsigned short*)(smem + OFF_AGGB);
  c.sG1   = (unsigned short*)(smem + OFF_G1);
  c.sT0   = (unsigned short*)(smem + OFF_T0);
  c.sXt   = (unsigned short*)(smem + OFF_XT);
  c.sH    = (unsigned short*)(smem + OFF_H);
  c.sState = (float*)(smem + OFF_STATE);
  c.wsInpT = wsInpT; c.wsHeadT = wsHeadT; c.masses = masses;
  c.tid = tid; c.wave = wave; c.lane = lane;

  // ---- one-time weight preload into LDS (bf16, [N][K]) ----
  for (int i = tid; i < 4 * LH_ * IH_; i += NTH_) sWih[i] = f2bf(Wih[i]);
  for (int i = tid; i < 4 * LH_ * IH_; i += NTH_) sWhh[i] = f2bf(Whh[i]);
  for (int i = tid; i < IH_ * GH_; i += NTH_) {
    int n = i >> 6, k = i & 63;
    sW1A[n * GH_ + k] = f2bf(iW1[(4 + k) * IH_ + n]);
  }
  for (int i = tid; i < 4 * IH_; i += NTH_) sW1S[i] = iW1[i];
  for (int i = tid; i < 4 * GH_; i += NTH_) sGW1[i] = gW1[i];
  for (int i = tid; i < GH_ * GH_; i += NTH_) {
    int n = i >> 6, k = i & 63;
    sGW2[n * GH_ + k] = f2bf(gW2[k * GH_ + n]);
  }
  for (int i = tid; i < IH_ * 4; i += NTH_) sHW2[i] = hW2[i];
  for (int i = tid; i < 64;  i += NTH_) { sB[BO_GB1+i] = gb1[i]; sB[BO_GB2+i] = gb2[i]; }
  for (int i = tid; i < 128; i += NTH_) {
    sB[BO_IB1+i] = ib1[i]; sB[BO_IB2+i] = ib2[i]; sB[BO_HB1+i] = hb1[i];
  }
  for (int i = tid; i < 512; i += NTH_) { sB[BO_BIH+i] = bih[i]; sB[BO_BHH+i] = bhh[i]; }
  if (tid < 4) sB[BO_HB2 + tid] = hb2[tid];
  for (int i = tid; i < MT_ * LH_; i += NTH_) c.sH[i] = 0;   // h0 = 0
  __syncthreads();

  // Hoist this wave's first 8 Whh B-fragments (K-blocks 0,1) into registers.
  v16bf whhF[8];
#pragma unroll
  for (int kk = 0; kk < 2; ++kk)
#pragma unroll
    for (int g = 0; g < 4; ++g)
      whhF[kk * 4 + g] = ldB(sWhh, LH_, g * LH_ + wave * 16, kk * 32, lane);

  v8f creg = Z8;                                             // c0 = 0 (in regs)

  // ---- encoder: 128 LSTM steps ----
#pragma unroll 1
  for (int t = 0; t < T_; ++t) {
    if (tid < MT_ * 4) {
      int m = tid >> 2, j = tid & 3;
      c.sState[m * 4 + j] = pastS[(((long)(b0 + m)) * T_ + t) * 4 + j];
    }
    __syncthreads();
    gnn_inp(c, pastP + (((long)b0) * T_ + t) * (P_ * 2), T_ * P_ * 2);
    lstm_step(c, creg, whhF);
  }

  // ---- autoregressive rollout: 30 steps ----
#pragma unroll 1
  for (int k = 0; k < F_; ++k) {
    // head layer1: 16x128, K=128, B streamed from transposed global copy
    {
      int n0 = wave * 16;
      v8f acc = Z8;
#pragma unroll
      for (int k0 = 0; k0 < LH_; k0 += 32)
        acc = wmma_bf16(ldA(c.sH, LH_, k0, lane),
                        ldB(wsHeadT, LH_, n0, k0, lane), acc);
      int n = n0 + (lane & 15);
      float bb = sB[BO_HB1 + n];
      int r0 = (lane & 16) ? 8 : 0;
#pragma unroll
      for (int v = 0; v < 8; ++v)
        c.sT0[(r0 + v) * IH_ + n] = f2bf(relu(acc[v] + bb));
    }
    __syncthreads();
    // head layer2 (128 -> 4), state update, emit prediction
    if (tid < MT_ * 4) {
      int m = tid >> 2, j = tid & 3;
      float d = sB[BO_HB2 + j];
#pragma unroll 4
      for (int kk = 0; kk < IH_; ++kk)
        d += bf2f(c.sT0[m * IH_ + kk]) * sHW2[kk * 4 + j];
      float ns = c.sState[m * 4 + j] + d;
      c.sState[m * 4 + j] = ns;
      out[(((long)(b0 + m)) * F_ + k) * 4 + j] = ns;
    }
    __syncthreads();
    gnn_inp(c, futP + (((long)b0) * F_ + k) * (P_ * 2), F_ * P_ * 2);
    lstm_step(c, creg, whhF);
  }
}

extern "C" void kernel_launch(void* const* d_in, const int* in_sizes, int n_in,
                              void* d_out, int out_size, void* d_ws, size_t ws_size,
                              hipStream_t stream) {
  (void)in_sizes; (void)n_in; (void)out_size; (void)ws_size;
  const float* pastP = (const float*)d_in[0];
  const float* pastS = (const float*)d_in[1];
  const float* futP  = (const float*)d_in[2];
  const float* mass  = (const float*)d_in[3];
  const float* gW1   = (const float*)d_in[4];
  const float* gb1   = (const float*)d_in[5];
  const float* gW2   = (const float*)d_in[6];
  const float* gb2   = (const float*)d_in[7];
  const float* iW1   = (const float*)d_in[8];
  const float* ib1   = (const float*)d_in[9];
  const float* iW2   = (const float*)d_in[10];
  const float* ib2   = (const float*)d_in[11];
  const float* Wih   = (const float*)d_in[12];
  const float* Whh   = (const float*)d_in[13];
  const float* bih   = (const float*)d_in[14];
  const float* bhh   = (const float*)d_in[15];
  const float* hW1   = (const float*)d_in[16];
  const float* hb1   = (const float*)d_in[17];
  const float* hW2   = (const float*)d_in[18];
  const float* hb2   = (const float*)d_in[19];

  unsigned short* wsInpT  = (unsigned short*)d_ws;          // [128][128] bf16
  unsigned short* wsHeadT = wsInpT + IH_ * IH_;             // [128][128] bf16

  asteroid_pack_weights<<<(IH_ * IH_ + 255) / 256, 256, 0, stream>>>(
      iW2, hW1, wsInpT, wsHeadT);

  asteroid_main<<<B_ / MT_, NTH_, SMEM_TOTAL, stream>>>(
      pastP, pastS, futP, mass, gW1, gb1, gW2, gb2, iW1, ib1, ib2,
      Wih, Whh, bih, bhh, hb1, hW2, hb2, wsInpT, wsHeadT, (float*)d_out);
}
